// CIDM_55138790146501
// MI455X (gfx1250) — compile-verified, matrix operand order checked by reading, not run on hardware
//
#include <hip/hip_runtime.h>
#include <hip/hip_bf16.h>
#include <cstddef>

// ---------------------------------------------------------------------------
// CDNA5 (gfx1250) fused LN -> QKV -> attention -> MLP, all GEMMs on
// v_wmma_f32_16x16x32_bf16 (wave32). Score row-block kept in 320KB LDS.
// All matrix operands arranged so every fragment load is a 16-byte b128 load.
// ---------------------------------------------------------------------------

#define HID   512
#define NSEQ  2048
#define NBAT  16
#define QSZ   1024   // 2*HID
#define BN    (NBAT * NSEQ)

typedef __attribute__((ext_vector_type(16))) __bf16 v16bf;
typedef __attribute__((ext_vector_type(8)))  __bf16 v8bf;
typedef __attribute__((ext_vector_type(4)))  __bf16 v4bf;
typedef __attribute__((ext_vector_type(8)))  float  v8f;

__device__ __forceinline__ v8f wmma_bf16(v16bf a, v16bf b, v8f c) {
    // 8 args: (neg_a, A, neg_b, B, c_mod, C, reuse_a, reuse_b)
    return __builtin_amdgcn_wmma_f32_16x16x32_bf16(
        false, a, false, b, (short)0, c, false, false);
}

// ISA 7.12.2 16-bit 16x32 operand layout: lane&15 = row; lanes 0-15 carry
// K in {0..7} U {16..23}, lanes 16-31 carry {8..15} U {24..31}; vgpr j holds
// pair (2j,2j+1). Per lane this is TWO CONTIGUOUS 16-byte runs -> 2x b128.
// Works for A-operands (rows = M) and for B-operands whose 32x16 tile is
// presented transposed (rows = N, k contiguous along the row).
__device__ __forceinline__ v16bf load_frag_row(const __bf16* rowBase,
                                               int rowStride, int lane) {
    const int r    = lane & 15;
    const int koff = (lane & 16) ? 8 : 0;
    const __bf16* p = rowBase + (size_t)r * rowStride + koff;
    v8bf lo = *(const v8bf*)(p);        // K = koff .. koff+7
    v8bf hi = *(const v8bf*)(p + 16);   // K = koff+16 .. koff+23
    return __builtin_shufflevector(lo, hi, 0, 1, 2, 3, 4, 5, 6, 7,
                                   8, 9, 10, 11, 12, 13, 14, 15);
}

// ---------------------------------------------------------------------------
// Kernel 0: f32 -> bf16 transposed weight conversion (src[rows][cols] ->
// dst[cols][rows]); run once per call, tiny.
// ---------------------------------------------------------------------------
__global__ void k_cvt_bf16_t(const float* __restrict__ src,
                             __bf16* __restrict__ dst, int rows, int cols) {
    const int n = rows * cols;
    for (int i = blockIdx.x * blockDim.x + threadIdx.x; i < n;
         i += gridDim.x * blockDim.x) {
        const int r = i / cols, c = i - r * cols;
        dst[(size_t)c * rows + r] = (__bf16)src[i];
    }
}

// ---------------------------------------------------------------------------
// Kernel 1: fused LayerNorm + QKV projection.
// Block = 256 threads (8 wave32), one 16-row token tile.
// Weights are bf16 TRANSPOSED ([n][k]) so B-fragments are b128 row loads.
// q,k written row-major via LDS-staged b128 stores; v written TRANSPOSED
// ([batch][hid][seq]) with one packed b128 store per C-tile per lane.
// ---------------------------------------------------------------------------
__global__ __launch_bounds__(256) void k_ln_qkv(
    const float* __restrict__ x, const float* __restrict__ ln_g,
    const float* __restrict__ ln_b,
    const __bf16* __restrict__ wqT, const __bf16* __restrict__ wkT,
    const __bf16* __restrict__ wvT,
    const float* __restrict__ bq, const float* __restrict__ bk,
    const float* __restrict__ bv,
    __bf16* __restrict__ qo, __bf16* __restrict__ ko,
    __bf16* __restrict__ vT) {
    __shared__ __bf16 xn[16 * HID];
    __shared__ __bf16 stg[8][16 * 16];   // per-wave C-tile staging
    __shared__ float redA[256], redB[256], s_mu[16], s_ri[16];

    const int tid  = threadIdx.x;
    const int lane = tid & 31;
    const int wave = tid >> 5;
    const size_t row0 = (size_t)blockIdx.x * 16;
    const int batch  = (int)(row0 >> 11);       // 2048 rows per batch
    const int rowInB = (int)(row0 & 2047);

    // ---- LayerNorm: 16 threads per row, float4 global loads ----
    const int r   = tid >> 4;
    const int c16 = tid & 15;
    const float4* xr4 = (const float4*)(x + (row0 + r) * HID);
    float s = 0.f, sq = 0.f;
#pragma unroll
    for (int i = 0; i < 8; ++i) {
        float4 v = xr4[c16 * 8 + i];
        s  += v.x + v.y + v.z + v.w;
        sq += v.x * v.x + v.y * v.y + v.z * v.z + v.w * v.w;
    }
    redA[tid] = s; redB[tid] = sq;
    __syncthreads();
    if (c16 == 0) {
        float ts = 0.f, tq = 0.f;
        for (int t = 0; t < 16; ++t) { ts += redA[r * 16 + t]; tq += redB[r * 16 + t]; }
        float m   = ts * (1.0f / HID);
        float var = tq * (1.0f / HID) - m * m;
        s_mu[r] = m;
        s_ri[r] = rsqrtf(var + 1e-5f);
    }
    __syncthreads();
    {
        const float m = s_mu[r], rv = s_ri[r];
        const float4* g4 = (const float4*)ln_g;
        const float4* b4 = (const float4*)ln_b;
        v4bf* xn4 = (v4bf*)(xn + r * HID);
#pragma unroll
        for (int i = 0; i < 8; ++i) {
            const int i4 = c16 * 8 + i;
            float4 v = xr4[i4], g = g4[i4], bb = b4[i4];
            v4bf o;
            o[0] = (__bf16)((v.x - m) * rv * g.x + bb.x);
            o[1] = (__bf16)((v.y - m) * rv * g.y + bb.y);
            o[2] = (__bf16)((v.z - m) * rv * g.z + bb.z);
            o[3] = (__bf16)((v.w - m) * rv * g.w + bb.w);
            xn4[i4] = o;
        }
    }
    __syncthreads();

    // ---- GEMM: 96 C tiles (q|k|v x 32 col-tiles), 12 per wave ----
    for (int i = 0; i < 12; ++i) {
        const int t   = wave * 12 + i;     // uniform within wave
        const int mm  = t >> 5;            // 0:q 1:k 2:v
        const int nb0 = (t & 31) * 16;
        const __bf16* WT; const float* bias;
        if (mm == 0)      { WT = wqT; bias = bq; }
        else if (mm == 1) { WT = wkT; bias = bk; }
        else              { WT = wvT; bias = bv; }
        v8f c = {};
        for (int kb = 0; kb < HID; kb += 32) {
            v16bf a = load_frag_row(xn + kb, HID, lane);
            v16bf b = load_frag_row(WT + (size_t)nb0 * HID + kb, HID, lane);
            c = wmma_bf16(a, b, c);
        }
        const float badd = bias[nb0 + (lane & 15)];
        const int rb = (lane & 16) ? 8 : 0;
        if (mm == 2) {
            // v: transposed store, 8 consecutive seq positions -> one b128
            v8bf pk;
#pragma unroll
            for (int e = 0; e < 8; ++e) pk[e] = (__bf16)(c[e] + badd);
            const int col = nb0 + (lane & 15);
            *(v8bf*)(vT + ((size_t)batch * HID + col) * NSEQ + rowInB + rb) = pk;
        } else {
            // q/k: stage tile in wave-private LDS, then b128 global stores
            __bf16* st = stg[wave];
#pragma unroll
            for (int e = 0; e < 8; ++e)
                st[(rb + e) * 16 + (lane & 15)] = (__bf16)(c[e] + badd);
            // wave-private LDS: DS ops are in-order, no barrier needed
            v8bf chunk = ((const v8bf*)st)[lane];      // row=lane>>1, half=lane&1
            __bf16* outp = (mm == 0) ? qo : ko;
            *(v8bf*)(outp + (row0 + (lane >> 1)) * HID + nb0 + (lane & 1) * 8) =
                chunk;
        }
    }
}

// ---------------------------------------------------------------------------
// Kernel 2: fused attention + MLP for one 16-query-row tile.
// Dynamic LDS: Sf f32[16][2048] (128KB) | Pb bf16[16][2048] (64KB) |
//              Qs bf16[16][512] (16KB, reused as O bf16) | red f32[512] (2KB)
// Total 215040 B < 320KB/WGP.
// ---------------------------------------------------------------------------
#define SMEM_SF 0
#define SMEM_PB (16 * NSEQ * 4)
#define SMEM_QS (SMEM_PB + 16 * NSEQ * 2)
#define SMEM_RED (SMEM_QS + 16 * HID * 2)
#define SMEM_TOTAL (SMEM_RED + 512 * 4)

__global__ __launch_bounds__(256) void k_attn_mlp(
    const __bf16* __restrict__ qbuf, const __bf16* __restrict__ kbuf,
    const __bf16* __restrict__ vT, const __bf16* __restrict__ w1T,
    const float* __restrict__ b1, const float* __restrict__ W2,
    const float* __restrict__ b2, float* __restrict__ out) {
    extern __shared__ char smem[];
    float*  Sf  = (float*)(smem + SMEM_SF);
    __bf16* Pb  = (__bf16*)(smem + SMEM_PB);
    __bf16* Qs  = (__bf16*)(smem + SMEM_QS);
    float*  red = (float*)(smem + SMEM_RED);

    const int tid  = threadIdx.x;
    const int lane = tid & 31;
    const int wave = tid >> 5;
    const int batch = blockIdx.x >> 7;   // 16 batches * 128 q-tiles
    const int qt    = blockIdx.x & 127;
    const size_t qrow0 = (size_t)batch * NSEQ + (size_t)qt * 16;

    // ---- stage Q tile into LDS (b128 copies) ----
    {
        const v8bf* src = (const v8bf*)(qbuf + qrow0 * HID);
        v8bf* dst = (v8bf*)Qs;
        for (int i = tid; i < 16 * HID / 8; i += 256) dst[i] = src[i];
    }
    __syncthreads();

    // ---- S = (Q K^T) * scale into LDS: 16 key-tiles per wave ----
    const float scale = 0.044194173824159216f;   // 1/sqrt(512)
    const __bf16* kbase = kbuf + (size_t)batch * NSEQ * HID;
    for (int i = 0; i < 16; ++i) {
        const int kt = wave + 8 * i;             // uniform within wave
        const __bf16* krows = kbase + (size_t)kt * 16 * HID;
        __builtin_prefetch(kbase + (size_t)(kt + 8) * 16 * HID, 0, 1);
        v8f c = {};
        for (int k0 = 0; k0 < HID; k0 += 32) {
            v16bf a = load_frag_row(Qs + k0, HID, lane);
            v16bf b = load_frag_row(krows + k0, HID, lane);  // B = K^T
            c = wmma_bf16(a, b, c);
        }
        const int rb  = (lane & 16) ? 8 : 0;
        const int col = kt * 16 + (lane & 15);
#pragma unroll
        for (int e = 0; e < 8; ++e)
            Sf[(size_t)(rb + e) * NSEQ + col] = c[e] * scale;
    }
    __syncthreads();

    // ---- exact softmax per row: 16 threads/row, float4 LDS accesses ----
    {
        const int r   = tid >> 4;
        const int c16 = tid & 15;
        float4* S4 = (float4*)(Sf + (size_t)r * NSEQ);
        float m = -3.4e38f;
#pragma unroll 4
        for (int i = 0; i < 32; ++i) {
            float4 v = S4[c16 + 16 * i];
            m = fmaxf(m, fmaxf(fmaxf(v.x, v.y), fmaxf(v.z, v.w)));
        }
        red[tid] = m;
        __syncthreads();
        if (c16 == 0) {
            float mm = red[r * 16];
            for (int t = 1; t < 16; ++t) mm = fmaxf(mm, red[r * 16 + t]);
            red[256 + r] = mm;
        }
        __syncthreads();
        const float mm = red[256 + r];
        float s = 0.f;
#pragma unroll 4
        for (int i = 0; i < 32; ++i) {
            float4 v = S4[c16 + 16 * i];
            v.x = __expf(v.x - mm); v.y = __expf(v.y - mm);
            v.z = __expf(v.z - mm); v.w = __expf(v.w - mm);
            S4[c16 + 16 * i] = v;
            s += v.x + v.y + v.z + v.w;
        }
        red[tid] = s;
        __syncthreads();
        if (c16 == 0) {
            float ss = 0.f;
            for (int t = 0; t < 16; ++t) ss += red[r * 16 + t];
            red[256 + r] = 1.0f / ss;
        }
        __syncthreads();
        const float inv = red[256 + r];
        v4bf* P4 = (v4bf*)(Pb + (size_t)r * NSEQ);
#pragma unroll 4
        for (int i = 0; i < 32; ++i) {
            float4 v = S4[c16 + 16 * i];
            v4bf p;
            p[0] = (__bf16)(v.x * inv); p[1] = (__bf16)(v.y * inv);
            p[2] = (__bf16)(v.z * inv); p[3] = (__bf16)(v.w * inv);
            P4[c16 + 16 * i] = p;
        }
    }
    __syncthreads();

    // ---- O = P @ V^T-stored : wave owns 64 output columns (4 C tiles) ----
    const __bf16* vtb = vT + (size_t)batch * HID * NSEQ;
    {
        v8f acc0 = {}, acc1 = {}, acc2 = {}, acc3 = {};
        const int nb0 = wave * 64;
        const __bf16* v0 = vtb + (size_t)(nb0)      * NSEQ;
        const __bf16* v1 = vtb + (size_t)(nb0 + 16) * NSEQ;
        const __bf16* v2 = vtb + (size_t)(nb0 + 32) * NSEQ;
        const __bf16* v3 = vtb + (size_t)(nb0 + 48) * NSEQ;
        for (int k0 = 0; k0 < NSEQ; k0 += 32) {
            v16bf a = load_frag_row(Pb + k0, NSEQ, lane);
            __builtin_prefetch(v0 + k0 + 32, 0, 1);
            acc0 = wmma_bf16(a, load_frag_row(v0 + k0, NSEQ, lane), acc0);
            acc1 = wmma_bf16(a, load_frag_row(v1 + k0, NSEQ, lane), acc1);
            acc2 = wmma_bf16(a, load_frag_row(v2 + k0, NSEQ, lane), acc2);
            acc3 = wmma_bf16(a, load_frag_row(v3 + k0, NSEQ, lane), acc3);
        }
        __bf16* Ob = Qs;                 // Q tile dead: reuse as bf16 O
        const int rb = (lane & 16) ? 8 : 0;
#pragma unroll
        for (int e = 0; e < 8; ++e) {
            const size_t ro = (size_t)(rb + e) * HID + nb0 + (lane & 15);
            Ob[ro]      = (__bf16)acc0[e];
            Ob[ro + 16] = (__bf16)acc1[e];
            Ob[ro + 32] = (__bf16)acc2[e];
            Ob[ro + 48] = (__bf16)acc3[e];
        }
    }
    __syncthreads();

    // ---- MLP: out = gelu(O@W1+b1) @ W2 + b2, fused reduction ----
    {
        const __bf16* Ob = Qs;
        float part[8] = {0, 0, 0, 0, 0, 0, 0, 0};
        for (int i = 0; i < 8; ++i) {
            const int jb = (wave + 8 * i) * 16;   // uniform within wave
            v8f c = {};
            for (int k0 = 0; k0 < HID; k0 += 32) {
                v16bf a = load_frag_row(Ob + k0, HID, lane);
                v16bf b = load_frag_row(w1T + (size_t)jb * HID + k0, HID, lane);
                c = wmma_bf16(a, b, c);
            }
            const int j = jb + (lane & 15);
            const float badd = b1[j];
            const float w2v  = W2[j];
#pragma unroll
            for (int e = 0; e < 8; ++e) {
                float h = c[e] + badd;
                float g = h / (1.0f + __expf(-1.702f * h));   // quick_gelu
                part[e] += g * w2v;
            }
        }
        // reduce over the 16 lanes of each half-wave (cols)
#pragma unroll
        for (int e = 0; e < 8; ++e) {
#pragma unroll
            for (int m = 8; m >= 1; m >>= 1)
                part[e] += __shfl_xor(part[e], m, 32);
        }
        if ((lane & 15) == 0) {
            const int rb = (lane & 16) ? 8 : 0;
#pragma unroll
            for (int e = 0; e < 8; ++e) red[wave * 16 + rb + e] = part[e];
        }
    }
    __syncthreads();
    if (tid < 16) {
        float acc = 0.f;
#pragma unroll
        for (int w = 0; w < 8; ++w) acc += red[w * 16 + tid];
        out[(size_t)batch * NSEQ + qt * 16 + tid] = acc + b2[0];
    }
}

// ---------------------------------------------------------------------------
// Host launcher. Workspace layout (bytes):
//   WqT/WkT/WvT bf16 (512KB each), W1T bf16 (1MB),
//   q bf16 [BN][HID] (32MB), k bf16 [BN][HID] (32MB),
//   vT bf16 [NBAT][HID][NSEQ] (32MB).
// ---------------------------------------------------------------------------
extern "C" void kernel_launch(void* const* d_in, const int* in_sizes, int n_in,
                              void* d_out, int out_size, void* d_ws,
                              size_t ws_size, hipStream_t stream) {
    const float* x    = (const float*)d_in[0];
    const float* ln_g = (const float*)d_in[1];
    const float* ln_b = (const float*)d_in[2];
    const float* Wq   = (const float*)d_in[3];
    const float* bq   = (const float*)d_in[4];
    const float* Wk   = (const float*)d_in[5];
    const float* bk   = (const float*)d_in[6];
    const float* Wv   = (const float*)d_in[7];
    const float* bv   = (const float*)d_in[8];
    const float* W1   = (const float*)d_in[9];
    const float* b1   = (const float*)d_in[10];
    const float* W2   = (const float*)d_in[11];
    const float* b2   = (const float*)d_in[12];
    float* out = (float*)d_out;

    char* ws = (char*)d_ws;
    const size_t OFF_WQ = 0;
    const size_t OFF_WK = OFF_WQ + (size_t)HID * HID * 2;
    const size_t OFF_WV = OFF_WK + (size_t)HID * HID * 2;
    const size_t OFF_W1 = OFF_WV + (size_t)HID * HID * 2;
    const size_t OFF_Q  = OFF_W1 + (size_t)HID * QSZ * 2;
    const size_t OFF_K  = OFF_Q + (size_t)BN * HID * 2;
    const size_t OFF_V  = OFF_K + (size_t)BN * HID * 2;

    __bf16* wqT = (__bf16*)(ws + OFF_WQ);
    __bf16* wkT = (__bf16*)(ws + OFF_WK);
    __bf16* wvT = (__bf16*)(ws + OFF_WV);
    __bf16* w1T = (__bf16*)(ws + OFF_W1);
    __bf16* qb  = (__bf16*)(ws + OFF_Q);
    __bf16* kb  = (__bf16*)(ws + OFF_K);
    __bf16* vt  = (__bf16*)(ws + OFF_V);

    k_cvt_bf16_t<<<256, 256, 0, stream>>>(Wq, wqT, HID, HID);
    k_cvt_bf16_t<<<256, 256, 0, stream>>>(Wk, wkT, HID, HID);
    k_cvt_bf16_t<<<256, 256, 0, stream>>>(Wv, wvT, HID, HID);
    k_cvt_bf16_t<<<256, 256, 0, stream>>>(W1, w1T, HID, QSZ);

    k_ln_qkv<<<BN / 16, 256, 0, stream>>>(x, ln_g, ln_b, wqT, wkT, wvT,
                                          bq, bk, bv, qb, kb, vt);

    k_attn_mlp<<<BN / 16, 256, SMEM_TOTAL, stream>>>(qb, kb, vt, w1T, b1,
                                                     W2, b2, out);
}
